// GraphAttnBias_49813030699100
// MI455X (gfx1250) — matrix operand order repaired and checked by reading, not compiled, not run on hardware
//
#include <hip/hip_runtime.h>

typedef __attribute__((ext_vector_type(16))) _Float16 v16h;
typedef __attribute__((ext_vector_type(8)))  _Float16 v8h;
typedef __attribute__((ext_vector_type(8)))  float    v8f;

#define WMMA_F16(a, b, c) \
  __builtin_amdgcn_wmma_f32_16x16x32_f16(false, (a), false, (b), (short)0, (c), false, false)

// ---------------------------------------------------------------------------
// Problem constants (from setup_inputs)
// ---------------------------------------------------------------------------
#define PB   4       // batch
#define PN   256     // nodes
#define PH   16      // heads
#define PD   5       // multi_hop_max_dist
#define PNE  1024    // n_edge per batch
#define PE   4096    // total edges
#define PL   20      // curve length
#define PEPS 1e-5f

// load a 16-half fragment (two contiguous 8-half runs) into a v16h
__device__ __forceinline__ v16h frag16(const _Float16* p0, const _Float16* p1) {
  const v8h lo = *(const v8h*)p0;
  const v8h hi = *(const v8h*)p1;
  v16h r;
#pragma unroll
  for (int i = 0; i < 8; ++i) { r[i] = lo[i]; r[8 + i] = hi[i]; }
  return r;
}

// ---------------------------------------------------------------------------
// misc small kernels
// ---------------------------------------------------------------------------
__global__ void zero_kernel(float* __restrict__ p, long n) {
  long i = (long)blockIdx.x * blockDim.x + threadIdx.x;
  if (i < n) p[i] = 0.f;
}

// pack f32 [N, Ktot] row-major -> f16 [N, Kpad] zero-padded
__global__ void pack_w_kernel(const float* __restrict__ w,
                              _Float16* __restrict__ wh, int Ktot, int Kpad) {
  int n = blockIdx.y;
  int k = blockIdx.x * blockDim.x + threadIdx.x;
  if (k >= Kpad) return;
  wh[(long)n * Kpad + k] = (_Float16)((k < Ktot) ? w[(long)n * Ktot + k] : 0.f);
}

// einsum B, transposed + padded: bt[n*96 + k] = (k<80) ? edis_w[k*16+n] : 0
__global__ void pack_bt_kernel(const float* __restrict__ edis_w,
                               _Float16* __restrict__ bt) {
  int idx = blockIdx.x * blockDim.x + threadIdx.x;
  if (idx >= 16 * 96) return;
  int n = idx / 96, k = idx % 96;
  bt[idx] = (_Float16)((k < PD * PH) ? edis_w[k * PH + n] : 0.f);
}

// ---------------------------------------------------------------------------
// Generic 2-layer MLP with BatchNorm (training stats) + ReLU, 3-pass.
// Reductions: wave32 shfl_xor butterfly -> shared stage -> global atomic.
// ---------------------------------------------------------------------------
__global__ void nl_pass_kernel(const float* __restrict__ x, int R, int Cin,
                               const float* __restrict__ W1,
                               const float* __restrict__ W2,
                               float* __restrict__ stats, int pass,
                               float* __restrict__ outbuf, int mode) {
  __shared__ float ss[32];
  int tid = threadIdx.x;
  if (pass != 3) {
    if (tid < 32) ss[tid] = 0.f;
    __syncthreads();
  }
  int r = blockIdx.x * blockDim.x + tid;
  bool active = (r < R);
  float pre1[16];
  if (active) {
    float xv[16];
    for (int c = 0; c < Cin; ++c) xv[c] = x[(long)r * Cin + c];
    for (int f = 0; f < 16; ++f) {
      float s = 0.f;
      for (int c = 0; c < Cin; ++c) s += W1[f * Cin + c] * xv[c];
      pre1[f] = s;
    }
  }
  if (pass == 1) {
    for (int f = 0; f < 16; ++f) {
      float s1 = active ? pre1[f] : 0.f;
      float s2 = active ? pre1[f] * pre1[f] : 0.f;
      for (int off = 16; off; off >>= 1) {
        s1 += __shfl_xor(s1, off, 32);
        s2 += __shfl_xor(s2, off, 32);
      }
      if ((tid & 31) == 0) {
        atomicAdd(&ss[f], s1);
        atomicAdd(&ss[16 + f], s2);
      }
    }
    __syncthreads();
    if (tid < 32) atomicAdd(&stats[tid], ss[tid]);
    return;
  }
  float pre2[16];
  if (active) {
    float h1[16];
    float invR = 1.f / (float)R;
    for (int f = 0; f < 16; ++f) {
      float mean = stats[f] * invR;
      float var  = stats[16 + f] * invR - mean * mean;
      float v = (pre1[f] - mean) * rsqrtf(var + PEPS);
      h1[f] = v > 0.f ? v : 0.f;
    }
    for (int k = 0; k < 16; ++k) {
      float s = 0.f;
      for (int f = 0; f < 16; ++f) s += W2[k * 16 + f] * h1[f];
      pre2[k] = s;
    }
  }
  if (pass == 2) {
    for (int k = 0; k < 16; ++k) {
      float s1 = active ? pre2[k] : 0.f;
      float s2 = active ? pre2[k] * pre2[k] : 0.f;
      for (int off = 16; off; off >>= 1) {
        s1 += __shfl_xor(s1, off, 32);
        s2 += __shfl_xor(s2, off, 32);
      }
      if ((tid & 31) == 0) {
        atomicAdd(&ss[k], s1);
        atomicAdd(&ss[16 + k], s2);
      }
    }
    __syncthreads();
    if (tid < 32) atomicAdd(&stats[32 + tid], ss[tid]);
    return;
  }
  // pass 3
  if (active) {
    float invR = 1.f / (float)R;
    for (int k = 0; k < 16; ++k) {
      float mean = stats[32 + k] * invR;
      float var  = stats[48 + k] * invR - mean * mean;
      float v = (pre2[k] - mean) * rsqrtf(var + PEPS);
      v = v > 0.f ? v : 0.f;
      if (mode == 0) {
        outbuf[(long)r * 16 + k] += v;
      } else {
        int NN = PN * PN;
        int bb = r / NN, ij = r % NN;
        outbuf[((long)bb * PH + k) * NN + ij] = v;
      }
    }
  }
}

// ---------------------------------------------------------------------------
// Implicit-GEMM Conv1d (kernel 3, pad 1) with WMMA f16->f32.  Weights arrive
// pre-packed f16 [O, Kpad] (zero-padded), so B fragments are 2 x b128 loads.
// A tile staged in LDS f16; fragments are 2 x ds b128 reads.  Input element
// type T (f32 for layer 1, f16 mirrors for layers 2/3).
// x viewed as [E, C, L] via strides (SE, SC, SL).  y: [E, O, L] f32.
// ---------------------------------------------------------------------------
template <typename T, int C, int O, int SE, int SC, int SL>
__global__ void conv_wmma_kernel(const T* __restrict__ x,
                                 const _Float16* __restrict__ wh,
                                 float* __restrict__ y) {
  constexpr int Ktot = C * 3;
  constexpr int Kpad = (Ktot + 31) & ~31;
  __shared__ _Float16 As[4][16 * Kpad];

  const int wave = threadIdx.x >> 5;
  const int lane = threadIdx.x & 31;
  const int rt   = blockIdx.x * 4 + wave;   // row tile (16 rows of (e,l))
  _Float16* Aw = &As[wave][0];

  // stage im2col A tile into LDS, zero-padded to Kpad
  for (int idx = lane; idx < 16 * Kpad; idx += 32) {
    int m = idx / Kpad, k = idx % Kpad;
    float val = 0.f;
    if (k < Ktot) {
      int row = rt * 16 + m;
      int e = row / PL, l = row % PL;
      int c = k / 3, kk = k % 3;
      int sl2 = l + kk - 1;
      if (sl2 >= 0 && sl2 < PL)
        val = (float)x[(long)e * SE + (long)c * SC + (long)sl2 * SL];
    }
    Aw[m * Kpad + k] = (_Float16)val;
  }
  __syncthreads();

  const int m    = lane & 15;
  const int kbA  = (lane < 16) ? 0 : 8;
  const int kbB  = (lane < 16) ? 0 : 8;
  const int nLan = lane & 15;

  for (int nt = 0; nt < O / 16; ++nt) {
    const int nc = nt * 16 + nLan;
    v8f acc = {};
#pragma unroll
    for (int k0 = 0; k0 < Kpad; k0 += 32) {
      v16h a = frag16(Aw + m * Kpad + k0 + kbA, Aw + m * Kpad + k0 + 16 + kbA);
      const _Float16* wp = wh + (long)nc * Kpad + k0 + kbB;
      v16h b = frag16(wp, wp + 16);
      acc = WMMA_F16(a, b, acc);
    }
#pragma unroll
    for (int v = 0; v < 8; ++v) {
      int mrow = (lane < 16) ? v : (8 + v);
      int row = rt * 16 + mrow;
      int e = row / PL, l = row % PL;
      y[((long)e * O + nc) * PL + l] = acc[v];
    }
  }
}

// per-channel BN stats over (E, L) for y: [E, O, L]; one block per channel.
__global__ void chan_stats_kernel(const float* __restrict__ y, int O,
                                  float* __restrict__ stats) {
  __shared__ float rs[256], rq[256];
  int o = blockIdx.x;
  const int n = PE * PL;
  float s = 0.f, q = 0.f;
  for (int i = threadIdx.x; i < n; i += blockDim.x) {
    int e = i / PL, l = i % PL;
    float v = y[((long)e * O + o) * PL + l];
    s += v; q += v * v;
  }
  rs[threadIdx.x] = s; rq[threadIdx.x] = q;
  __syncthreads();
  for (int st = blockDim.x >> 1; st > 0; st >>= 1) {
    if (threadIdx.x < st) {
      rs[threadIdx.x] += rs[threadIdx.x + st];
      rq[threadIdx.x] += rq[threadIdx.x + st];
    }
    __syncthreads();
  }
  if (threadIdx.x == 0) {
    float mean = rs[0] / (float)n;
    float var  = rq[0] / (float)n - mean * mean;
    stats[o * 2]     = mean;
    stats[o * 2 + 1] = rsqrtf(var + PEPS);
  }
}

// BN + LeakyReLU.  hout!=null: write f16 mirror only (next consumer is WMMA);
// hout==null: write f32 in place.
__global__ void bn_lrelu_kernel(float* __restrict__ y, int O,
                                const float* __restrict__ stats,
                                _Float16* __restrict__ hout) {
  long idx = (long)blockIdx.x * blockDim.x + threadIdx.x;
  long total = (long)PE * O * PL;
  if (idx >= total) return;
  int o = (int)((idx / PL) % O);
  float v = (y[idx] - stats[o * 2]) * stats[o * 2 + 1];
  v = v >= 0.f ? v : 0.01f * v;
  if (hout) hout[idx] = (_Float16)v;
  else      y[idx] = v;
}

__global__ void pool_kernel(const float* __restrict__ y3,
                            _Float16* __restrict__ poolh) {
  long idx = (long)blockIdx.x * blockDim.x + threadIdx.x;  // e*256 + o
  if (idx >= (long)PE * 256) return;
  float s = 0.f;
  const float* p = y3 + idx * PL;
#pragma unroll
  for (int l = 0; l < PL; ++l) s += p[l];
  poolh[idx] = (_Float16)(s * (1.f / PL));
}

// ---------------------------------------------------------------------------
// All-f16 GEMM with WMMA: C[M,N] = A[M,KPAD] @ W[N,KPAD]^T, f32 accumulate.
// A and W pre-packed f16 (zero-padded); every fragment = 2 x b128 loads.
// 4 waves / block, one 16x16 tile per wave.
// ---------------------------------------------------------------------------
template <int KPAD, int N>
__global__ void gemm_h_kernel(const _Float16* __restrict__ Ah,
                              const _Float16* __restrict__ Wh,
                              float* __restrict__ Cb, int M) {
  constexpr int TILES_N = N >> 4;
  int lane = threadIdx.x & 31;
  int wave = threadIdx.x >> 5;
  int tile = blockIdx.x * 4 + wave;
  int mt = tile / TILES_N, nt = tile % TILES_N;
  if (mt * 16 >= M) return;  // wave-uniform

  const int m    = lane & 15;
  const long row = (long)(mt * 16 + m);
  const int kbA  = (lane < 16) ? 0 : 8;
  const int kbB  = (lane < 16) ? 0 : 8;
  const int nCol = nt * 16 + (lane & 15);

  v8f acc = {};
#pragma unroll
  for (int k0 = 0; k0 < KPAD; k0 += 32) {
    const _Float16* ap = Ah + row * KPAD + k0;
    v16h a = frag16(ap + kbA, ap + 16 + kbA);
    const _Float16* wp = Wh + (long)nCol * KPAD + k0 + kbB;
    v16h b = frag16(wp, wp + 16);
    acc = WMMA_F16(a, b, acc);
  }
#pragma unroll
  for (int v = 0; v < 8; ++v) {
    int r = mt * 16 + ((lane < 16) ? v : 8 + v);
    Cb[(long)r * N + nCol] = acc[v];
  }
}

// per-column BN stats for buf: [R, C] row-major; one block per column.
__global__ void col_stats_kernel(const float* __restrict__ buf, int R, int C,
                                 float* __restrict__ stats) {
  __shared__ float rs[256], rq[256];
  int c = blockIdx.x;
  float s = 0.f, q = 0.f;
  for (int r = threadIdx.x; r < R; r += blockDim.x) {
    float v = buf[(long)r * C + c];
    s += v; q += v * v;
  }
  rs[threadIdx.x] = s; rq[threadIdx.x] = q;
  __syncthreads();
  for (int st = blockDim.x >> 1; st > 0; st >>= 1) {
    if (threadIdx.x < st) {
      rs[threadIdx.x] += rs[threadIdx.x + st];
      rq[threadIdx.x] += rq[threadIdx.x + st];
    }
    __syncthreads();
  }
  if (threadIdx.x == 0) {
    float mean = rs[0] / (float)R;
    float var  = rq[0] / (float)R - mean * mean;
    stats[c * 2]     = mean;
    stats[c * 2 + 1] = rsqrtf(var + PEPS);
  }
}

// column BN; act: 0 none, 1 relu; dst!=null -> dst += v else buf = v;
// hdst!=null -> also write f16 mirror.
__global__ void bn_col_apply_kernel(float* __restrict__ buf, int R, int C,
                                    const float* __restrict__ stats, int act,
                                    float* __restrict__ dst,
                                    _Float16* __restrict__ hdst) {
  long idx = (long)blockIdx.x * blockDim.x + threadIdx.x;
  if (idx >= (long)R * C) return;
  int c = (int)(idx % C);
  float v = (buf[idx] - stats[c * 2]) * stats[c * 2 + 1];
  if (act && v < 0.f) v = 0.f;
  if (dst) dst[idx] += v;
  else     buf[idx] = v;
  if (hdst) hdst[idx] = (_Float16)v;
}

// aggh[e,h] = f16(proj[src[e],h] + proj[dst[e],h] + ef[e,h]); [E,32] K-padded
__global__ void agg_kernel(const float* __restrict__ proj,
                           const int* __restrict__ src,
                           const int* __restrict__ dst,
                           const float* __restrict__ ef,
                           _Float16* __restrict__ aggh) {
  long idx = (long)blockIdx.x * blockDim.x + threadIdx.x;
  if (idx >= (long)PE * 32) return;
  int e = (int)(idx >> 5), h = (int)(idx & 31);
  float v = 0.f;
  if (h < 16)
    v = proj[(long)src[e] * PH + h] + proj[(long)dst[e] * PH + h] + ef[(long)e * PH + h];
  aggh[idx] = (_Float16)v;
}

// ---------------------------------------------------------------------------
// Multi-hop einsum fused with gather:  per 16 (b,i,j) rows,
//   C[16x16] = Agather[16 x 96] @ Bt[16 x 96]^T   (K=D*H=80, padded to 96)
// ef2h: f16 [E+1, 16] with an all-zero row at index PE (padding slots and the
// K-pad hop redirect there) -> A fragment = 2 unconditional b128 loads.
// bt: f16 [16, 96] pre-transposed/padded -> B fragment = 2 b128 loads.
// Epilogue: branchless scale via the hardware reciprocal (v_rcp_f32).
// ---------------------------------------------------------------------------
__global__ void einsum_wmma_kernel(const _Float16* __restrict__ ef2h,
                                   const int* __restrict__ epath,
                                   const int* __restrict__ sdist,
                                   const _Float16* __restrict__ bt,
                                   float* __restrict__ ebias) {
  int lane = threadIdx.x & 31;
  int wave = threadIdx.x >> 5;
  int t0 = (blockIdx.x * 4 + wave) * 16;

  const int m = lane & 15;
  const long t = (long)t0 + m;
  const int NN = PN * PN;
  const int b = (int)(t / NN);

  long base[6];
#pragma unroll
  for (int d = 0; d < PD; ++d) {
    int e = epath[t * PD + d];
    base[d] = (long)((e < PNE) ? (b * PNE + e) : PE) * PH;  // zero row at PE
  }
  base[5] = (long)PE * PH;  // K-pad hop -> zero row

  const int hb   = (lane < 16) ? 0 : 8;
  const int kbB  = (lane < 16) ? 0 : 8;
  const int nCol = lane & 15;

  v8f acc = {};
#pragma unroll
  for (int k0 = 0; k0 < 96; k0 += 32) {
    const int d0 = k0 >> 4;
    v16h a = frag16(ef2h + base[d0] + hb, ef2h + base[d0 + 1] + hb);
    const _Float16* bp = bt + (long)nCol * 96 + k0 + kbB;
    v16h bf = frag16(bp, bp + 16);
    acc = WMMA_F16(a, bf, acc);
  }

#pragma unroll
  for (int v = 0; v < 8; ++v) {
    long tr = (long)t0 + ((lane < 16) ? v : 8 + v);
    int s = sdist[tr];
    int sd_ = (s == 0) ? 1 : s;
    if (sd_ > 1) sd_ -= 1;
    if (sd_ > PD) sd_ = PD;
    // branchless: v_cvt + v_add + v_rcp_f32 (1 ulp; denom in [1+eps, 5+eps])
    float scale = __builtin_amdgcn_rcpf((float)sd_ + 1e-6f);
    float val = acc[v] * scale;
    if (!(val == val)) val = 0.f;                     // NaN -> 0
    if (val >  3.0e38f) val =  3.0e38f;               // +inf clamp
    if (val < -3.0e38f) val = -3.0e38f;               // -inf clamp
    ebias[tr * PH + nCol] = val;
  }
}

// ---------------------------------------------------------------------------
// Final assembly into [B, H, N+1, N+1]
// ---------------------------------------------------------------------------
__global__ void assemble_kernel(const float* __restrict__ attn,
                                const int* __restrict__ sdist,
                                const float* __restrict__ sd_emb,
                                const float* __restrict__ virt,
                                const float* __restrict__ ang,
                                const float* __restrict__ cen,
                                const float* __restrict__ eb,
                                float* __restrict__ out) {
  const int Np = PN + 1;
  long total = (long)PB * PH * Np * Np;
  long idx = (long)blockIdx.x * blockDim.x + threadIdx.x;
  if (idx >= total) return;
  int q = (int)(idx % Np);
  long t1 = idx / Np;
  int p = (int)(t1 % Np);
  long t2 = t1 / Np;
  int h = (int)(t2 % PH);
  int b = (int)(t2 / PH);

  float val = 2.f * attn[((long)b * Np + p) * Np + q];  // broadcast + final add
  if (p > 0 && q > 0) {
    int i = p - 1, j = q - 1;
    long t = ((long)b * PN + i) * PN + j;
    int s = sdist[t];
    val += sd_emb[(long)s * PH + h];
    long NN = (long)PN * PN;
    val += ang[((long)b * PH + h) * NN + (long)i * PN + j];
    val += cen[((long)b * PH + h) * NN + (long)i * PN + j];
    val += eb[t * PH + h];
  }
  if (p == 0 || q == 0) val += virt[h];
  out[idx] = val;
}

// ---------------------------------------------------------------------------
// host side
// ---------------------------------------------------------------------------
extern "C" void kernel_launch(void* const* d_in, const int* in_sizes, int n_in,
                              void* d_out, int out_size, void* d_ws, size_t ws_size,
                              hipStream_t stream) {
  (void)in_sizes; (void)n_in; (void)out_size; (void)ws_size;

  const float* attn_bias = (const float*)d_in[0];
  const int*   sdist     = (const int*)  d_in[1];
  const float* angle     = (const float*)d_in[2];
  const float* centd     = (const float*)d_in[3];
  const int*   epath     = (const int*)  d_in[4];
  const float* edata     = (const float*)d_in[5];
  const float* etype     = (const float*)d_in[6];
  const float* elen      = (const float*)d_in[7];
  const float* econv     = (const float*)d_in[8];
  const int*   srcI      = (const int*)  d_in[10];
  const int*   dstI      = (const int*)  d_in[11];
  const float* nodef     = (const float*)d_in[12];
  const float* sdemb     = (const float*)d_in[13];
  const float* virt      = (const float*)d_in[14];
  const float* angle_W1  = (const float*)d_in[15];
  const float* angle_W2  = (const float*)d_in[16];
  const float* cent_W1   = (const float*)d_in[17];
  const float* cent_W2   = (const float*)d_in[18];
  const float* etype_W1  = (const float*)d_in[19];
  const float* etype_W2  = (const float*)d_in[20];
  const float* elen_W1   = (const float*)d_in[21];
  const float* elen_W2   = (const float*)d_in[22];
  const float* econv_W1  = (const float*)d_in[23];
  const float* econv_W2  = (const float*)d_in[24];
  const float* curv_c1   = (const float*)d_in[25];
  const float* curv_c2   = (const float*)d_in[26];
  const float* curv_c3   = (const float*)d_in[27];
  const float* curv_fc   = (const float*)d_in[28];
  const float* edis_w    = (const float*)d_in[29];
  const float* ec_Wp     = (const float*)d_in[30];
  const float* ec_W1     = (const float*)d_in[31];
  const float* ec_W2     = (const float*)d_in[32];
  float* out = (float*)d_out;
  float* ws  = (float*)d_ws;

  // ---- f32 workspace (floats; every block 16-float aligned) ----
  const long SZ_BHNN = (long)PB * PH * PN * PN;        // 4,194,304
  float* STATS = ws;                                   // 2048 used
  float* ANG   = ws + 4096;
  float* CEN   = ANG + SZ_BHNN;
  float* EB    = CEN + SZ_BHNN;                        // [B*N*N, H]
  float* Y1    = EB + SZ_BHNN;                         // [E,64,20] raw conv1
  float* Y2    = Y1 + (long)PE * 64 * PL;              // [E,128,20] raw conv2
  float* Y3    = Y2 + (long)PE * 128 * PL;             // [E,256,20] conv3/bn
  float* CUR   = Y3 + (long)PE * 256 * PL;             // [E,16]
  float* EF    = CUR + (long)PE * PH;                  // [E,16] accumulator
  float* PROJ  = EF + (long)PE * PH;                   // [B*N,16]
  float* HED   = PROJ + (long)PB * PN * PH;            // [E,64]
  float* EF2   = HED + (long)PE * 64;                  // [E,16]

  // ---- f16 workspace (halves; every block 16-half aligned) ----
  _Float16* hwsBase = (_Float16*)(EF2 + (long)PE * PH);
  _Float16* WH_C1  = hwsBase;                          // [64, 64]
  _Float16* WH_C2  = WH_C1 + 4096;                     // [128, 192]
  _Float16* WH_C3  = WH_C2 + 24576;                    // [256, 384]
  _Float16* WH_FC  = WH_C3 + 98304;                    // [16, 256]
  _Float16* WH_WP  = WH_FC + 4096;                     // [16, 256]
  _Float16* WH_E1  = WH_WP + 4096;                     // [64, 32]
  _Float16* WH_E2  = WH_E1 + 2048;                     // [16, 64]
  _Float16* BT     = WH_E2 + 1024;                     // [16, 96]
  _Float16* NODEFH = BT + 1536;                        // [1024, 256]
  _Float16* POOLH  = NODEFH + (long)PB * PN * 256;     // [E, 256]
  _Float16* AGGH   = POOLH + (long)PE * 256;           // [E, 32] K-padded
  _Float16* HEDH   = AGGH + (long)PE * 32;             // [E, 64]
  _Float16* EF2H   = HEDH + (long)PE * 64;             // [E+1, 16] (+zero row)
  _Float16* Y1H    = EF2H + (long)(PE + 1) * PH;       // [E,64,20]
  _Float16* Y2H    = Y1H + (long)PE * 64 * PL;         // [E,128,20]

  // stats sub-offsets
  float* ST_ANG = STATS + 0;
  float* ST_CEN = STATS + 64;
  float* ST_ETY = STATS + 128;
  float* ST_ELN = STATS + 192;
  float* ST_ECV = STATS + 256;
  float* ST_C1  = STATS + 320;   // 64 ch * 2
  float* ST_C2  = STATS + 448;   // 128 ch * 2
  float* ST_C3  = STATS + 704;   // 256 ch * 2
  float* ST_FC  = STATS + 1216;  // 16 * 2
  float* ST_E1  = STATS + 1248;  // 64 * 2
  float* ST_E2  = STATS + 1376;  // 16 * 2

  zero_kernel<<<(2048 + 255) / 256, 256, 0, stream>>>(STATS, 2048);
  zero_kernel<<<((long)PE * PH + 255) / 256, 256, 0, stream>>>(EF, (long)PE * PH);
  zero_kernel<<<1, 256, 0, stream>>>((float*)(EF2H + (long)PE * PH), PH / 2);  // zero pad row

  // ---- pack all WMMA operand buffers to f16 (once per launch) ----
  pack_w_kernel<<<dim3(1, 64),   128, 0, stream>>>(curv_c1, WH_C1, 36, 64);
  pack_w_kernel<<<dim3(2, 128),  128, 0, stream>>>(curv_c2, WH_C2, 192, 192);
  pack_w_kernel<<<dim3(3, 256),  128, 0, stream>>>(curv_c3, WH_C3, 384, 384);
  pack_w_kernel<<<dim3(2, 16),   128, 0, stream>>>(curv_fc, WH_FC, 256, 256);
  pack_w_kernel<<<dim3(2, 16),   128, 0, stream>>>(ec_Wp,   WH_WP, 256, 256);
  pack_w_kernel<<<dim3(1, 64),   128, 0, stream>>>(ec_W1,   WH_E1, 16, 32);
  pack_w_kernel<<<dim3(1, 16),   128, 0, stream>>>(ec_W2,   WH_E2, 64, 64);
  pack_w_kernel<<<dim3(2, 1024), 128, 0, stream>>>(nodef,   NODEFH, 256, 256);
  pack_bt_kernel<<<(16 * 96 + 127) / 128, 128, 0, stream>>>(edis_w, BT);

  // --- angle / centroid branches (R = B*N*N, Cin = 1, transpose to [B,H,N,N])
  const int Rbig = PB * PN * PN;
  dim3 gBig((Rbig + 255) / 256);
  for (int p = 1; p <= 3; ++p)
    nl_pass_kernel<<<gBig, 256, 0, stream>>>(angle, Rbig, 1, angle_W1, angle_W2,
                                             ST_ANG, p, ANG, 1);
  for (int p = 1; p <= 3; ++p)
    nl_pass_kernel<<<gBig, 256, 0, stream>>>(centd, Rbig, 1, cent_W1, cent_W2,
                                             ST_CEN, p, CEN, 1);

  // --- edge scalar branches accumulate into EF  (R = E)
  dim3 gE((PE + 255) / 256);
  for (int p = 1; p <= 3; ++p)
    nl_pass_kernel<<<gE, 256, 0, stream>>>(etype, PE, 11, etype_W1, etype_W2,
                                           ST_ETY, p, EF, 0);
  for (int p = 1; p <= 3; ++p)
    nl_pass_kernel<<<gE, 256, 0, stream>>>(elen, PE, 1, elen_W1, elen_W2,
                                           ST_ELN, p, EF, 0);
  for (int p = 1; p <= 3; ++p)
    nl_pass_kernel<<<gE, 256, 0, stream>>>(econv, PE, 3, econv_W1, econv_W2,
                                           ST_ECV, p, EF, 0);

  // --- curve encoder: conv(12->64) -> bn+lrelu -> conv(64->128) -> ... -> pool
  const int convBlocks = (PE * PL / 16) / 4;  // 1280 blocks, 4 row tiles each
  conv_wmma_kernel<float, 12, 64, 240, 1, 12>
      <<<convBlocks, 128, 0, stream>>>(edata, WH_C1, Y1);
  chan_stats_kernel<<<64, 256, 0, stream>>>(Y1, 64, ST_C1);
  bn_lrelu_kernel<<<((long)PE * 64 * PL + 255) / 256, 256, 0, stream>>>(Y1, 64, ST_C1, Y1H);

  conv_wmma_kernel<_Float16, 64, 128, 64 * PL, PL, 1>
      <<<convBlocks, 128, 0, stream>>>(Y1H, WH_C2, Y2);
  chan_stats_kernel<<<128, 256, 0, stream>>>(Y2, 128, ST_C2);
  bn_lrelu_kernel<<<((long)PE * 128 * PL + 255) / 256, 256, 0, stream>>>(Y2, 128, ST_C2, Y2H);

  conv_wmma_kernel<_Float16, 128, 256, 128 * PL, PL, 1>
      <<<convBlocks, 128, 0, stream>>>(Y2H, WH_C3, Y3);
  chan_stats_kernel<<<256, 256, 0, stream>>>(Y3, 256, ST_C3);
  bn_lrelu_kernel<<<((long)PE * 256 * PL + 255) / 256, 256, 0, stream>>>(Y3, 256, ST_C3, nullptr);

  pool_kernel<<<((long)PE * 256 + 255) / 256, 256, 0, stream>>>(Y3, POOLH);

  // curve fc: [E,256] @ [16,256]^T -> CUR; BN (no activation); EF += CUR
  gemm_h_kernel<256, 16><<<(PE / 16) / 4, 128, 0, stream>>>(POOLH, WH_FC, CUR, PE);
  col_stats_kernel<<<16, 256, 0, stream>>>(CUR, PE, 16, ST_FC);
  bn_col_apply_kernel<<<((long)PE * 16 + 255) / 256, 256, 0, stream>>>(
      CUR, PE, 16, ST_FC, 0, EF, nullptr);

  // --- EdgeConv: proj(node_feat); agg; MLP 16->64 (bn,relu); 64->16; bn,relu
  gemm_h_kernel<256, 16><<<((PB * PN) / 16) / 4, 128, 0, stream>>>(NODEFH, WH_WP, PROJ, PB * PN);
  agg_kernel<<<((long)PE * 32 + 255) / 256, 256, 0, stream>>>(PROJ, srcI, dstI, EF, AGGH);
  gemm_h_kernel<32, 64><<<((PE / 16) * 4) / 4, 128, 0, stream>>>(AGGH, WH_E1, HED, PE);
  col_stats_kernel<<<64, 256, 0, stream>>>(HED, PE, 64, ST_E1);
  bn_col_apply_kernel<<<((long)PE * 64 + 255) / 256, 256, 0, stream>>>(
      HED, PE, 64, ST_E1, 1, nullptr, HEDH);
  gemm_h_kernel<64, 16><<<(PE / 16) / 4, 128, 0, stream>>>(HEDH, WH_E2, EF2, PE);
  col_stats_kernel<<<16, 256, 0, stream>>>(EF2, PE, 16, ST_E2);
  bn_col_apply_kernel<<<((long)PE * 16 + 255) / 256, 256, 0, stream>>>(
      EF2, PE, 16, ST_E2, 1, nullptr, EF2H);

  // --- fused gather + per-hop einsum (WMMA), scaling, nan_to_num
  einsum_wmma_kernel<<<((PB * PN * PN) / 16) / 4, 128, 0, stream>>>(EF2H, epath, sdist, BT, EB);

  // --- final assembly
  {
    long total = (long)PB * PH * (PN + 1) * (PN + 1);
    assemble_kernel<<<(total + 255) / 256, 256, 0, stream>>>(
        attn_bias, sdist, sdemb, virt, ANG, CEN, EB, out);
  }
}
